// PointSetEmbedding_16509854286647
// MI455X (gfx1250) — matrix operand order repaired and checked by reading, not compiled, not run on hardware
//
#include <hip/hip_runtime.h>
#include <hip/hip_bf16.h>

#define B_   8
#define N_   8192
#define D_   64
#define S_   1024
#define NS_  32
#define R2_  0.04f

typedef _Float16 half_t;
typedef __attribute__((ext_vector_type(16))) _Float16 v16h;
typedef __attribute__((ext_vector_type(8)))  _Float16 v8h;
typedef __attribute__((ext_vector_type(8)))  float    v8f;

union V16U { v16h v; v8h h[2]; half_t e[16]; };

#define FEAT_P 104   // padded f16 row stride for 96 feature cols (208B, 16B aligned)
#define H1_P   136   // padded f16 row stride for 128 hidden cols (272B, 16B aligned)

#define W1H_ELEMS (128 * 96)
#define W2H_ELEMS (256 * 128)

__device__ __forceinline__ float silu_f(float x) {
  return x / (1.0f + __expf(-x));
}

// ---- one-shot weight conversion: f32 -> f16, W1 zero-padded 66 -> 96 cols ----
__global__ __launch_bounds__(256)
void prep_weights(const float* __restrict__ W1, const float* __restrict__ W2,
                  half_t* __restrict__ W1h, half_t* __restrict__ W2h) {
  const int i = blockIdx.x * 256 + threadIdx.x;
  if (i < W1H_ELEMS) {
    const int o = i / 96;
    const int c = i - o * 96;
    W1h[i] = (c < 66) ? (half_t)W1[o * 66 + c] : (half_t)0.0f;
  }
  if (i < W2H_ELEMS) {
    W2h[i] = (half_t)W2[i];
  }
}

// load f16 [p..p+7] and [p+16..p+23] (16B aligned) as a WMMA B fragment
__device__ __forceinline__ v16h load_frag16(const half_t* p) {
  V16U u;
  u.h[0] = *(const v8h*)p;
  u.h[1] = *(const v8h*)(p + 16);
  return u.v;
}

__global__ __launch_bounds__(128)
void pse_fused(const float* __restrict__ xyz, const float* __restrict__ pts,
               const half_t* __restrict__ W1h, const float* __restrict__ b1,
               const half_t* __restrict__ W2h, const float* __restrict__ b2,
               float* __restrict__ out) {
  __shared__ half_t featLDS[NS_ * FEAT_P];
  __shared__ half_t h1LDS[NS_ * H1_P];
  __shared__ int    idxLDS[NS_];
  __shared__ float  accLDS[256];

  const int s    = blockIdx.x;
  const int b    = blockIdx.y;
  const int tid  = threadIdx.x;
  const int lane = tid & 31;
  const int wid  = tid >> 5;

  const long bxy = (long)b * 2 * N_;
  const long bpt = (long)b * D_ * N_;

  // zero the cross-wave reduction accumulator
  accLDS[tid]       = 0.0f;
  accLDS[tid + 128] = 0.0f;

  // ---------------- ball query: first 32 in-ball indices, ascending (wave 0) ----------------
  if (wid == 0) {
    const float qx = xyz[bxy + s];
    const float qy = xyz[bxy + N_ + s];
    int count = 0;
    for (int base = 0; base < N_ && count < NS_; base += 32) {
      const int n = base + lane;
      const float dx = xyz[bxy + n] - qx;
      const float dy = xyz[bxy + N_ + n] - qy;
      const bool inb = (dx * dx + dy * dy) <= R2_;
      const unsigned mask = __builtin_amdgcn_ballot_w32(inb);
      const int pos = count + __popc(mask & ((1u << lane) - 1u));
      if (inb && pos < NS_) idxLDS[pos] = n;
      count += __popc(mask);
    }
    const int cnt = count < NS_ ? count : NS_;
    // pad short groups with the first index (query point itself is always in-ball)
    if (lane >= cnt) idxLDS[lane] = idxLDS[0];
  }
  __syncthreads();

  // ---------------- gather + stage 32x96 feature tile as f16 in LDS ----------------
  {
    const int r = tid >> 2;          // row 0..31
    const int q = tid & 3;           // 24-col quarter
    const int n = idxLDS[r];
    const float qx = xyz[bxy + s];
    const float qy = xyz[bxy + N_ + s];
    half_t* dst = &featLDS[r * FEAT_P];
    #pragma unroll
    for (int j = 0; j < 24; ++j) {
      const int c = q * 24 + j;
      float v;
      if (c == 0)      v = xyz[bxy + n] - qx;
      else if (c == 1) v = xyz[bxy + N_ + n] - qy;
      else if (c < 66) v = pts[bpt + (long)(c - 2) * N_ + n];
      else             v = 0.0f;
      dst[c] = (half_t)v;
    }
  }
  __syncthreads();

  const int mt   = wid & 1;        // M-tile (rows 16*mt .. 16*mt+15)
  const int oh   = wid >> 1;       // output-channel half
  const int l15  = lane & 15;
  const int hs   = lane >> 4;      // A/B half-lane select
  const int rowA = mt * 16 + l15;

  // ---------------- GEMM1: feat(16x96) x W1^T(96x64) per wave ----------------
  v16h a1[3];
  #pragma unroll
  for (int kt = 0; kt < 3; ++kt) {
    const half_t* p = &featLDS[rowA * FEAT_P + kt * 32 + hs * 8];
    V16U u;
    u.h[0] = *(const v8h*)p;          // K = base .. base+7
    u.h[1] = *(const v8h*)(p + 16);   // K = base+16 .. base+23
    a1[kt] = u.v;
  }

  #pragma unroll
  for (int nt = 0; nt < 4; ++nt) {
    const int colg = oh * 64 + nt * 16 + l15;      // output channel 0..127
    const half_t* wrow = W1h + colg * 96;
    v8f acc = {};
    #pragma unroll
    for (int kt = 0; kt < 3; ++kt) {
      const int cs = kt * 32 + hs * 8;
      acc = __builtin_amdgcn_wmma_f32_16x16x32_f16(false, a1[kt], false,
                                                   load_frag16(wrow + cs),
                                                   (short)0, acc, false, false);
    }
    const float bias = b1[colg];
    #pragma unroll
    for (int i = 0; i < 8; ++i) {
      const int row = mt * 16 + hs * 8 + i;
      h1LDS[row * H1_P + colg] = (half_t)silu_f(acc[i] + bias);
    }
  }
  __syncthreads();

  // ---------------- GEMM2: H1(16x128) x W2^T(128x128) per wave ----------------
  v16h a2[4];
  #pragma unroll
  for (int kt = 0; kt < 4; ++kt) {
    const half_t* p = &h1LDS[rowA * H1_P + kt * 32 + hs * 8];
    V16U u;
    u.h[0] = *(const v8h*)p;
    u.h[1] = *(const v8h*)(p + 16);
    a2[kt] = u.v;
  }

  #pragma unroll
  for (int nt = 0; nt < 8; ++nt) {
    const int colg = oh * 128 + nt * 16 + l15;     // output channel 0..255
    const half_t* wrow = W2h + colg * 128;
    v8f acc = {};
    #pragma unroll
    for (int kt = 0; kt < 4; ++kt) {
      const int cs = kt * 32 + hs * 8;
      acc = __builtin_amdgcn_wmma_f32_16x16x32_f16(false, a2[kt], false,
                                                   load_frag16(wrow + cs),
                                                   (short)0, acc, false, false);
    }
    const float bias = b2[colg];
    float sum8 = 0.0f;
    #pragma unroll
    for (int i = 0; i < 8; ++i) sum8 += silu_f(acc[i] + bias);
    atomicAdd(&accLDS[colg], sum8);                // ds_add_f32, 8-row partial
  }
  __syncthreads();

  // ---------------- mean over the 32 samples, lane stride of S_ floats ----------------
  {
    const float invn = 1.0f / 32.0f;
    int o2 = tid;
    out[((long)b * 256 + o2) * S_ + s] = accLDS[o2] * invn;
    o2 = tid + 128;
    out[((long)b * 256 + o2) * S_ + s] = accLDS[o2] * invn;
  }
}

extern "C" void kernel_launch(void* const* d_in, const int* in_sizes, int n_in,
                              void* d_out, int out_size, void* d_ws, size_t ws_size,
                              hipStream_t stream) {
  const float* xyz = (const float*)d_in[0];
  const float* pts = (const float*)d_in[1];
  const float* W1  = (const float*)d_in[2];
  const float* b1  = (const float*)d_in[3];
  const float* W2  = (const float*)d_in[4];
  const float* b2  = (const float*)d_in[5];
  float* out = (float*)d_out;

  half_t* W1h = (half_t*)d_ws;                       // 24576 B
  half_t* W2h = (half_t*)((char*)d_ws + W1H_ELEMS * sizeof(half_t));  // 65536 B

  // convert weights to f16 (W1 zero-padded to 96 cols) in workspace
  prep_weights<<<dim3((W2H_ELEMS + 255) / 256), dim3(256), 0, stream>>>(W1, W2, W1h, W2h);

  dim3 grid(S_, B_);
  dim3 block(128);
  pse_fused<<<grid, block, 0, stream>>>(xyz, pts, W1h, b1, W2h, b2, out);
}
